// AnswerSelection_64261300683081
// MI455X (gfx1250) — compile-verified, matrix operand order checked by reading, not run on hardware
//
#include <hip/hip_runtime.h>
#include <math.h>

// ---------------- problem constants ----------------
#define B_   8
#define Q_   256
#define A_   128
#define E_   256
#define H_   256     // 2*HID
#define HID_ 128
#define G4_  512     // 4*HID
#define F_   256
#define F3_  768     // 3*F

typedef __attribute__((ext_vector_type(16))) __bf16 v16bf;
typedef __attribute__((ext_vector_type(8)))  float  v8f;

// ---------------------------------------------------------------------------
// Embedding gather + f32 -> bf16 convert.  out[bt*E + e] = bf16(emb[tok[bt]*E + e])
// ---------------------------------------------------------------------------
__global__ void embed_bf16_kernel(const int* __restrict__ tok,
                                  const float* __restrict__ emb,
                                  __bf16* __restrict__ out, int BT) {
  int i = blockIdx.x * blockDim.x + threadIdx.x;
  if (i >= BT * E_) return;
  int e  = i & (E_ - 1);
  int bt = i >> 8;                 // E_ == 256
  out[i] = (__bf16)emb[(size_t)tok[bt] * E_ + e];
}

// f32 -> bf16 convert (for weight matrices)
__global__ void cvt_bf16_kernel(const float* __restrict__ in,
                                __bf16* __restrict__ out, int n) {
  int i = blockIdx.x * blockDim.x + threadIdx.x;
  if (i < n) out[i] = (__bf16)in[i];
}

// ---------------------------------------------------------------------------
// WMMA GEMM:  C[M,N] = A[M,K] @ Bm[N,K]^T + bias0[N] + bias1[N]
// A, Bm bf16 row-major; C f32 row-major.  M%16==0, N%64==0, K%32==0.
// blockDim = 128 (4 waves); each wave owns one 16x16 C tile.
// Fragment packing follows CDNA5 ISA 7.12.2 (16-bit A 16x32 layout):
//   lane<16 : rows m0+lane,  k elements {k0..k0+7, k0+16..k0+23}
//   lane>=16: rows m0+lane-16, k elements {k0+8..k0+15, k0+24..k0+31}
// ---------------------------------------------------------------------------
__global__ void gemm_bf16_wmma_kernel(const __bf16* __restrict__ A,
                                      const __bf16* __restrict__ Bm,
                                      const float* __restrict__ bias0,
                                      const float* __restrict__ bias1,
                                      float* __restrict__ C,
                                      int M, int N, int K) {
  const int wave = threadIdx.x >> 5;
  const int lane = threadIdx.x & 31;
  const int m0 = blockIdx.x * 16;
  const int n0 = (blockIdx.y * 4 + wave) * 16;

  const int row  = lane & 15;
  const int koff = (lane >> 4) * 8;

  const __bf16* pa = A  + (size_t)(m0 + row) * K + koff;
  const __bf16* pb = Bm + (size_t)(n0 + row) * K + koff;

  v8f acc = {};
  for (int k0 = 0; k0 < K; k0 += 32) {
    __builtin_prefetch(pa + 128, 0, 1);   // global_prefetch_b8 (next k-iters)
    __builtin_prefetch(pb + 128, 0, 1);
    union { uint4 u[2]; v16bf v; } fa, fb;
    fa.u[0] = *(const uint4*)(pa);        // 8 bf16: k0+koff   .. +7
    fa.u[1] = *(const uint4*)(pa + 16);   // 8 bf16: k0+16+koff.. +7
    fb.u[0] = *(const uint4*)(pb);
    fb.u[1] = *(const uint4*)(pb + 16);
    acc = __builtin_amdgcn_wmma_f32_16x16x32_bf16(
        /*neg_a=*/false, fa.v, /*neg_b=*/false, fb.v,
        /*c_mod=*/(short)0, acc, /*reuse_a=*/false, /*reuse_b=*/false);
    pa += 32;
    pb += 32;
  }

  // C/D layout: lanes 0-15 -> M = m0 + r, N = n0+lane; lanes 16-31 -> M = m0+8+r.
  const int n     = n0 + (lane & 15);
  const int mbase = m0 + ((lane >> 4) ? 8 : 0);
  const float bv  = bias0[n] + bias1[n];
#pragma unroll
  for (int r = 0; r < 8; ++r)
    C[(size_t)(mbase + r) * N + n] = acc[r] + bv;
}

// ---------------------------------------------------------------------------
// One LSTM direction.  pre: [B, T, 512] precomputed x@w_ih^T + (b_ih+b_hh).
// Block per batch element, 512 threads; h,c live in LDS; sequential over T.
// out[b,t, dir_off + j] = h_j(t)   (out is [B,T,256])
// ---------------------------------------------------------------------------
__global__ void lstm_dir_kernel(const float* __restrict__ pre,
                                const float* __restrict__ w_hh, // [512,128]
                                float* __restrict__ out,
                                int T, int reverse, int dir_off) {
  const int b = blockIdx.x;
  const int j = threadIdx.x;            // 0..511
  __shared__ float h[HID_], c[HID_], g[G4_];
  if (j < HID_) { h[j] = 0.f; c[j] = 0.f; }
  __syncthreads();

  const float* w = w_hh + (size_t)j * HID_;
  for (int s = 0; s < T; ++s) {
    const int t = reverse ? (T - 1 - s) : s;
    const float* p = pre + ((size_t)b * T + t) * G4_;
    float acc = p[j];
#pragma unroll 8
    for (int k = 0; k < HID_; ++k) acc = fmaf(h[k], w[k], acc);
    g[j] = acc;
    __syncthreads();
    if (j < HID_) {
      float iv = 1.f / (1.f + __expf(-g[j]));
      float fv = 1.f / (1.f + __expf(-g[HID_ + j]));
      float gv = tanhf(g[2 * HID_ + j]);
      float ov = 1.f / (1.f + __expf(-g[3 * HID_ + j]));
      float cn = fv * c[j] + iv * gv;
      float hn = ov * tanhf(cn);
      c[j] = cn;
      h[j] = hn;
      out[((size_t)b * T + t) * H_ + dir_off + j] = hn;
    }
    __syncthreads();
  }
}

// ---------------------------------------------------------------------------
// Coattention.  q view: q[b,h,qi] = qlstm_flat[b*H*Q + h*Q + qi] (reshape trick).
// Cq[b,h,qi] = sum_a softmax_a(q*a) * a ;  Ca[b,h,ai] = sum_q softmax_q(q*a) * q
// ---------------------------------------------------------------------------
__global__ void coatt_cq_kernel(const float* __restrict__ qf,
                                const float* __restrict__ af,
                                float* __restrict__ Cq) {
  int idx = blockIdx.x * blockDim.x + threadIdx.x;   // B*H*Q
  if (idx >= B_ * H_ * Q_) return;
  int qi = idx % Q_;
  int h  = (idx / Q_) % H_;
  int b  = idx / (Q_ * H_);
  float qv = qf[idx];
  const float* av = af + ((size_t)b * H_ + h) * A_;
  float mx = -3.4e38f;
  for (int a = 0; a < A_; ++a) mx = fmaxf(mx, qv * av[a]);
  float den = 0.f, num = 0.f;
  for (int a = 0; a < A_; ++a) {
    float e = __expf(qv * av[a] - mx);
    den += e;
    num += e * av[a];
  }
  Cq[idx] = num / den;
  (void)qi;
}

__global__ void coatt_ca_kernel(const float* __restrict__ qf,
                                const float* __restrict__ af,
                                float* __restrict__ Ca) {
  int idx = blockIdx.x * blockDim.x + threadIdx.x;   // B*H*A
  if (idx >= B_ * H_ * A_) return;
  int h = (idx / A_) % H_;
  int b = idx / (A_ * H_);
  float av = af[idx];
  const float* qv = qf + ((size_t)b * H_ + h) * Q_;
  float mx = -3.4e38f;
  for (int q = 0; q < Q_; ++q) mx = fmaxf(mx, qv[q] * av);
  float den = 0.f, num = 0.f;
  for (int q = 0; q < Q_; ++q) {
    float e = __expf(qv[q] * av - mx);
    den += e;
    num += e * qv[q];
  }
  Ca[idx] = num / den;
}

// ---------------------------------------------------------------------------
// conv1d (OIH weights) -> global max over time -> tanh.  Block per (f, b),
// 128 threads strided over output positions; LDS max reduction.
// out[b, branch_off + f]
// ---------------------------------------------------------------------------
__global__ void conv_max_tanh_kernel(const float* __restrict__ X,   // [B,H,T]
                                     const float* __restrict__ W,   // [F,H,k]
                                     const float* __restrict__ bias,
                                     float* __restrict__ out,       // [B,3F]
                                     int T, int k, int pad, int branch_off) {
  const int f = blockIdx.x;
  const int b = blockIdx.y;
  const int To = T + 2 * pad - k + 1;
  float best = -3.4e38f;
  for (int to = threadIdx.x; to < To; to += blockDim.x) {
    float acc = bias[f];
    for (int h = 0; h < H_; ++h) {
      const float* xr = X + ((size_t)b * H_ + h) * T;
      const float* wr = W + ((size_t)f * H_ + h) * k;
      for (int j = 0; j < k; ++j) {
        int t = to + j - pad;
        if (t >= 0 && t < T) acc = fmaf(wr[j], xr[t], acc);
      }
    }
    best = fmaxf(best, acc);
  }
  __shared__ float red[128];
  red[threadIdx.x] = best;
  __syncthreads();
  for (int s = 64; s > 0; s >>= 1) {
    if (threadIdx.x < s)
      red[threadIdx.x] = fmaxf(red[threadIdx.x], red[threadIdx.x + s]);
    __syncthreads();
  }
  if (threadIdx.x == 0)
    out[(size_t)b * F3_ + branch_off + f] = tanhf(red[0]);
}

// ---------------------------------------------------------------------------
// Cosine similarity per batch row:  out[b] = <qo,ao> / max(|qo||ao|, 1e-8)
// ---------------------------------------------------------------------------
__global__ void cosine_kernel(const float* __restrict__ qo,
                              const float* __restrict__ ao,
                              float* __restrict__ out) {
  const int b = blockIdx.x;
  const int t = threadIdx.x;
  float d = 0.f, nq = 0.f, na = 0.f;
  for (int i = t; i < F3_; i += 256) {
    float x = qo[(size_t)b * F3_ + i];
    float y = ao[(size_t)b * F3_ + i];
    d += x * y; nq += x * x; na += y * y;
  }
  __shared__ float sd[256], sq[256], sa[256];
  sd[t] = d; sq[t] = nq; sa[t] = na;
  __syncthreads();
  for (int s = 128; s > 0; s >>= 1) {
    if (t < s) { sd[t] += sd[t + s]; sq[t] += sq[t + s]; sa[t] += sa[t + s]; }
    __syncthreads();
  }
  if (t == 0)
    out[b] = sd[0] / fmaxf(sqrtf(sq[0]) * sqrtf(sa[0]), 1e-8f);
}

// ---------------------------------------------------------------------------
extern "C" void kernel_launch(void* const* d_in, const int* in_sizes, int n_in,
                              void* d_out, int out_size, void* d_ws, size_t ws_size,
                              hipStream_t stream) {
  const int*   question = (const int*)  d_in[0];
  const int*   answer   = (const int*)  d_in[1];
  const float* emb      = (const float*)d_in[2];
  const float* w_ih_f   = (const float*)d_in[3];
  const float* w_hh_f   = (const float*)d_in[4];
  const float* b_ih_f   = (const float*)d_in[5];
  const float* b_hh_f   = (const float*)d_in[6];
  const float* w_ih_b   = (const float*)d_in[7];
  const float* w_hh_b   = (const float*)d_in[8];
  const float* b_ih_b   = (const float*)d_in[9];
  const float* b_hh_b   = (const float*)d_in[10];
  const float* conv_w1  = (const float*)d_in[11];
  const float* conv_b1  = (const float*)d_in[12];
  const float* conv_w2  = (const float*)d_in[13];
  const float* conv_b2  = (const float*)d_in[14];
  const float* conv_w3  = (const float*)d_in[15];
  const float* conv_b3  = (const float*)d_in[16];
  float* out = (float*)d_out;

  // --- workspace carve-up (256B aligned slices) ---
  char* ws = (char*)d_ws;
  size_t off = 0;
  auto take = [&](size_t bytes) -> char* {
    char* p = ws + off;
    off += (bytes + 255) & ~(size_t)255;
    return p;
  };
  __bf16* xq_bf  = (__bf16*)take((size_t)B_ * Q_ * E_ * 2);
  __bf16* xa_bf  = (__bf16*)take((size_t)B_ * A_ * E_ * 2);
  __bf16* wf_bf  = (__bf16*)take((size_t)G4_ * E_ * 2);
  __bf16* wb_bf  = (__bf16*)take((size_t)G4_ * E_ * 2);
  float*  preq_f = (float*) take((size_t)B_ * Q_ * G4_ * 4);
  float*  preq_b = (float*) take((size_t)B_ * Q_ * G4_ * 4);
  float*  prea_f = (float*) take((size_t)B_ * A_ * G4_ * 4);
  float*  prea_b = (float*) take((size_t)B_ * A_ * G4_ * 4);
  float*  qlstm  = (float*) take((size_t)B_ * Q_ * H_ * 4);
  float*  alstm  = (float*) take((size_t)B_ * A_ * H_ * 4);
  float*  cq     = (float*) take((size_t)B_ * H_ * Q_ * 4);
  float*  ca     = (float*) take((size_t)B_ * H_ * A_ * 4);
  float*  qo     = (float*) take((size_t)B_ * F3_ * 4);
  float*  ao     = (float*) take((size_t)B_ * F3_ * 4);
  (void)ws_size; (void)n_in; (void)in_sizes; (void)out_size;

  // 1) embeddings -> bf16 ; weights -> bf16
  {
    int nq = B_ * Q_ * E_, na = B_ * A_ * E_, nw = G4_ * E_;
    embed_bf16_kernel<<<(nq + 255) / 256, 256, 0, stream>>>(question, emb, xq_bf, B_ * Q_);
    embed_bf16_kernel<<<(na + 255) / 256, 256, 0, stream>>>(answer,   emb, xa_bf, B_ * A_);
    cvt_bf16_kernel<<<(nw + 255) / 256, 256, 0, stream>>>(w_ih_f, wf_bf, nw);
    cvt_bf16_kernel<<<(nw + 255) / 256, 256, 0, stream>>>(w_ih_b, wb_bf, nw);
  }

  // 2) input projections via WMMA:  pre = x @ w_ih^T + (b_ih + b_hh)
  {
    dim3 blk(128);
    dim3 gq(B_ * Q_ / 16, G4_ / 64);   // 128 x 8
    dim3 ga(B_ * A_ / 16, G4_ / 64);   //  64 x 8
    gemm_bf16_wmma_kernel<<<gq, blk, 0, stream>>>(xq_bf, wf_bf, b_ih_f, b_hh_f, preq_f, B_ * Q_, G4_, E_);
    gemm_bf16_wmma_kernel<<<gq, blk, 0, stream>>>(xq_bf, wb_bf, b_ih_b, b_hh_b, preq_b, B_ * Q_, G4_, E_);
    gemm_bf16_wmma_kernel<<<ga, blk, 0, stream>>>(xa_bf, wf_bf, b_ih_f, b_hh_f, prea_f, B_ * A_, G4_, E_);
    gemm_bf16_wmma_kernel<<<ga, blk, 0, stream>>>(xa_bf, wb_bf, b_ih_b, b_hh_b, prea_b, B_ * A_, G4_, E_);
  }

  // 3) sequential BiLSTM recurrences (fwd + bwd per sequence)
  lstm_dir_kernel<<<B_, G4_, 0, stream>>>(preq_f, w_hh_f, qlstm, Q_, 0, 0);
  lstm_dir_kernel<<<B_, G4_, 0, stream>>>(preq_b, w_hh_b, qlstm, Q_, 1, HID_);
  lstm_dir_kernel<<<B_, G4_, 0, stream>>>(prea_f, w_hh_f, alstm, A_, 0, 0);
  lstm_dir_kernel<<<B_, G4_, 0, stream>>>(prea_b, w_hh_b, alstm, A_, 1, HID_);

  // 4) channel-wise coattention (flat-view reshape semantics)
  coatt_cq_kernel<<<(B_ * H_ * Q_) / 256, 256, 0, stream>>>(qlstm, alstm, cq);
  coatt_ca_kernel<<<(B_ * H_ * A_) / 256, 256, 0, stream>>>(qlstm, alstm, ca);

  // 5) conv branches -> global max -> tanh -> concat
  {
    dim3 blk(128);
    dim3 g(F_, B_);
    conv_max_tanh_kernel<<<g, blk, 0, stream>>>(cq, conv_w1, conv_b1, qo, Q_, 1, 0, 0);
    conv_max_tanh_kernel<<<g, blk, 0, stream>>>(cq, conv_w2, conv_b2, qo, Q_, 3, 2, F_);
    conv_max_tanh_kernel<<<g, blk, 0, stream>>>(cq, conv_w3, conv_b3, qo, Q_, 5, 2, 2 * F_);
    conv_max_tanh_kernel<<<g, blk, 0, stream>>>(ca, conv_w1, conv_b1, ao, A_, 1, 0, 0);
    conv_max_tanh_kernel<<<g, blk, 0, stream>>>(ca, conv_w2, conv_b2, ao, A_, 3, 2, F_);
    conv_max_tanh_kernel<<<g, blk, 0, stream>>>(ca, conv_w3, conv_b3, ao, A_, 5, 2, 2 * F_);
  }

  // 6) cosine similarity
  cosine_kernel<<<B_, 256, 0, stream>>>(qo, ao, out);
}
